// DiffAttention_13752485281904
// MI455X (gfx1250) — matrix operand order repaired and checked by reading, not compiled, not run on hardware
//
#include <hip/hip_runtime.h>
#include <hip/hip_bf16.h>

#define DIM 1024
#define NH 16
#define HDIM 32
#define BATCH 4
#define SEQ 4096
#define LAMBDA_INIT 0.2f
#define RMS_EPS 1e-5f

typedef __attribute__((ext_vector_type(16))) __bf16 v16bf;
typedef __attribute__((ext_vector_type(8)))  float  v8f;
typedef __attribute__((ext_vector_type(4)))  unsigned int v4u;
typedef __attribute__((ext_vector_type(8)))  int v8i;
typedef __attribute__((ext_vector_type(4)))  int v4i;

#if __has_builtin(__builtin_amdgcn_tensor_load_to_lds) && \
    __has_builtin(__builtin_amdgcn_s_wait_tensorcnt)
#define HAVE_TDM 1
#else
#define HAVE_TDM 0
#endif

static __device__ inline v8f zero8() {
  v8f z;
#pragma unroll
  for (int i = 0; i < 8; ++i) z[i] = 0.0f;
  return z;
}

static __device__ inline v8f wmma_bf16(v16bf a, v16bf b, v8f c) {
  // D(f32 16x16) = A(bf16 16x32) * B(bf16 32x16) + C
  return __builtin_amdgcn_wmma_f32_16x16x32_bf16(false, a, false, b, (short)0, c,
                                                 false, false);
}

// ---- DPP16 row (16-lane) reductions: rotate-by-{1,2,4,8} butterfly, pure VALU
#define DPP_ROR_F32(x, n)                                                     \
  __builtin_bit_cast(float, __builtin_amdgcn_update_dpp(                      \
      0, __builtin_bit_cast(int, (x)), 0x120 + (n), 0xF, 0xF, true))

static __device__ inline float row_max16(float x) {
  x = fmaxf(x, DPP_ROR_F32(x, 1));
  x = fmaxf(x, DPP_ROR_F32(x, 2));
  x = fmaxf(x, DPP_ROR_F32(x, 4));
  x = fmaxf(x, DPP_ROR_F32(x, 8));
  return x;
}
static __device__ inline float row_sum16(float x) {
  x += DPP_ROR_F32(x, 1);
  x += DPP_ROR_F32(x, 2);
  x += DPP_ROR_F32(x, 4);
  x += DPP_ROR_F32(x, 8);
  return x;
}

// A-fragment (16x32 bf16), per ISA layout:
//   lanes 0-15 : VGPR0-3 K=0..7,  VGPR4-7 K=16..23   (row = lane)
//   lanes16-31 : VGPR0-3 K=8..15, VGPR4-7 K=24..31   (row = lane-16)
static __device__ inline v16bf frag_a(const __bf16* tile, int stride) {
  const int lane = threadIdx.x & 31, lo = lane & 15, hi = lane >> 4;
  v16bf r;
  ((int4*)&r)[0] = *(const int4*)(tile + lo * stride + hi * 8);
  ((int4*)&r)[1] = *(const int4*)(tile + lo * stride + 16 + hi * 8);
  return r;
}

// B-fragment (32x16 bf16): lane n holds column n; lanes 0-15 K=0..7,16..23,
// lanes 16-31 K=8..15,24..31, 2 K per VGPR. LDS layout [col][k].
static __device__ inline v16bf frag_b(const __bf16* tile, int stride) {
  const int lane = threadIdx.x & 31, lo = lane & 15, hi = lane >> 4;
  const __bf16* p = tile + lo * stride + hi * 8;
  v16bf r;
  ((int4*)&r)[0] = *(const int4*)(p);
  ((int4*)&r)[1] = *(const int4*)(p + 16);
  return r;
}

static __device__ inline int4 pack_bf16x8(float4 a, float4 b) {
  union { int4 i; __bf16 h[8]; } u;
  u.h[0] = (__bf16)a.x; u.h[1] = (__bf16)a.y; u.h[2] = (__bf16)a.z; u.h[3] = (__bf16)a.w;
  u.h[4] = (__bf16)b.x; u.h[5] = (__bf16)b.y; u.h[6] = (__bf16)b.z; u.h[7] = (__bf16)b.w;
  return u.i;
}

// Fetch 16 consecutive values as packed bf16 (2x int4).
template <bool SRC_BF16>
static __device__ inline void fetch16(const void* p, long off, int4* s) {
  if (SRC_BF16) {
    const __bf16* q = (const __bf16*)p + off;
    s[0] = ((const int4*)q)[0];
    s[1] = ((const int4*)q)[1];
  } else {
    const float* q = (const float*)p + off;
    float4 f0 = ((const float4*)q)[0];
    float4 f1 = ((const float4*)q)[1];
    float4 f2 = ((const float4*)q)[2];
    float4 f3 = ((const float4*)q)[3];
    s[0] = pack_bf16x8(f0, f1);
    s[1] = pack_bf16x8(f2, f3);
  }
}

// ---------------------------------------------------------------------------
// C[M,N] = A[M,K] * B[N,K]^T (+bias). 128x128x32 tiles, 8 waves, bf16 WMMA.
// ---------------------------------------------------------------------------
template <bool A_BF16, bool OUT_F32_BIAS>
__global__ __launch_bounds__(256) void gemm_wmma_kernel(
    const void* __restrict__ Ap, const float* __restrict__ B,
    const float* __restrict__ bias, void* __restrict__ Cp,
    int M, int N, int K) {
  __shared__ alignas(16) __bf16 As[128 * 32];
  __shared__ alignas(16) __bf16 Bs[128 * 32];

  const int tid  = threadIdx.x;
  const int row  = tid >> 1;   // 0..127
  const int half = tid & 1;    // k sub-offset 0/16
  const int m0 = blockIdx.x * 128;
  const int n0 = blockIdx.y * 128;

  const int wave = tid >> 5;
  const int wm = wave & 1;     // 2 waves along M (64 rows each)
  const int wn = wave >> 1;    // 4 waves along N (32 cols each)

  v8f acc[4][2];
#pragma unroll
  for (int i = 0; i < 4; ++i)
#pragma unroll
    for (int j = 0; j < 2; ++j) acc[i][j] = zero8();

  const int nk = K >> 5;
  int4 sa[2], sb[2];
  fetch16<A_BF16>(Ap, (long)(m0 + row) * K + half * 16, sa);
  fetch16<false>(B, (long)(n0 + row) * K + half * 16, sb);

  for (int kt = 0; kt < nk; ++kt) {
    *(int4*)(As + row * 32 + half * 16)     = sa[0];
    *(int4*)(As + row * 32 + half * 16 + 8) = sa[1];
    *(int4*)(Bs + row * 32 + half * 16)     = sb[0];
    *(int4*)(Bs + row * 32 + half * 16 + 8) = sb[1];
    __syncthreads();
    if (kt + 1 < nk) {  // prefetch next K-slice while computing
      fetch16<A_BF16>(Ap, (long)(m0 + row) * K + (kt + 1) * 32 + half * 16, sa);
      fetch16<false>(B, (long)(n0 + row) * K + (kt + 1) * 32 + half * 16, sb);
    }
    if (kt + 2 < nk) {  // L2 prefetch two slices ahead (global_prefetch_b8)
      if (A_BF16)
        __builtin_prefetch((const __bf16*)Ap + (long)(m0 + row) * K + (kt + 2) * 32, 0, 1);
      else
        __builtin_prefetch((const float*)Ap + (long)(m0 + row) * K + (kt + 2) * 32, 0, 1);
      __builtin_prefetch(B + (long)(n0 + row) * K + (kt + 2) * 32, 0, 1);
    }
    v16bf af[4], bf[2];
#pragma unroll
    for (int t = 0; t < 4; ++t) af[t] = frag_a(As + (wm * 64 + t * 16) * 32, 32);
#pragma unroll
    for (int t = 0; t < 2; ++t) bf[t] = frag_b(Bs + (wn * 32 + t * 16) * 32, 32);
#pragma unroll
    for (int i = 0; i < 4; ++i)
#pragma unroll
      for (int j = 0; j < 2; ++j) acc[i][j] = wmma_bf16(af[i], bf[j], acc[i][j]);
    __syncthreads();
  }

  const int lane = tid & 31, lo = lane & 15, hi = lane >> 4;
#pragma unroll
  for (int i = 0; i < 4; ++i)
#pragma unroll
    for (int j = 0; j < 2; ++j) {
      const long gmBase = m0 + wm * 64 + i * 16 + hi * 8;
      const long gn = n0 + wn * 32 + j * 16 + lo;
#pragma unroll
      for (int r = 0; r < 8; ++r) {
        const long gm = gmBase + r;
        const float v = acc[i][j][r];
        if (OUT_F32_BIAS) {
          ((float*)Cp)[gm * N + gn] = v + bias[gn];
        } else {
          ((__bf16*)Cp)[gm * N + gn] = (__bf16)v;
        }
      }
    }
}

// ---------------------------------------------------------------------------
// lambda = exp(sum(lq1*lk1)) - exp(sum(lq2*lk2)) + lambda_init
// ---------------------------------------------------------------------------
__global__ void lambda_kernel(const float* __restrict__ q1, const float* __restrict__ k1,
                              const float* __restrict__ q2, const float* __restrict__ k2,
                              float* __restrict__ out) {
  const int i = threadIdx.x;  // 32 threads, HDIM == 32
  float s1 = q1[i] * k1[i];
  float s2 = q2[i] * k2[i];
#pragma unroll
  for (int m = 16; m >= 1; m >>= 1) {
    s1 += __shfl_xor(s1, m, 32);
    s2 += __shfl_xor(s2, m, 32);
  }
  if (i == 0) out[0] = __expf(s1) - __expf(s2) + LAMBDA_INIT;
}

// ---------------------------------------------------------------------------
// Differential flash attention. Block = (b, h, 128 query rows); each of 8
// waves owns 16 query rows. qkv layout: [B*N, 3072] bf16; per row:
//   q1 @ h*64, q2 @ h*64+32, k1 @ 1024+h*64, k2 @ 1024+h*64+32, v @ 2048+h*64.
// K tile is staged via the Tensor Data Mover when available; V tile is staged
// transposed by hand (TDM cannot transpose). Writes o (diff + RMS-norm, *0.8)
// as bf16 [B*N, 1024].
// ---------------------------------------------------------------------------
__global__ __launch_bounds__(256) void diffattn_kernel(
    const __bf16* __restrict__ qkv, const float* __restrict__ lam_p,
    const float* __restrict__ sub_w, __bf16* __restrict__ obuf) {
  __shared__ alignas(16) __bf16 Klds[32 * 64];      // [key][dim0..63] (k1|k2)
  __shared__ alignas(16) __bf16 Vlds[64 * 32];      // [e][key]  (transposed)
  __shared__ alignas(16) __bf16 Pst[8][2][16 * 32]; // per-wave P staging (C->A)

  const int tid = threadIdx.x;
  const int wave = tid >> 5;
  const int lane = tid & 31, lo = lane & 15, hi = lane >> 4;
  const int b = blockIdx.y >> 4;
  const int h = blockIdx.y & 15;
  const int q0 = blockIdx.x * 128 + wave * 16;
  const float scale = 0.1767766952966369f;  // 32^-0.5
  const float lam = lam_p[0];

  // Q fragments straight from global; fold the 1/sqrt(hd) scale into Q once.
  v16bf Q1, Q2;
  {
    const __bf16* qb = qkv + (long)(b * SEQ + q0 + lo) * 3072 + h * 64;
    ((int4*)&Q1)[0] = *(const int4*)(qb + hi * 8);
    ((int4*)&Q1)[1] = *(const int4*)(qb + 16 + hi * 8);
    ((int4*)&Q2)[0] = *(const int4*)(qb + 32 + hi * 8);
    ((int4*)&Q2)[1] = *(const int4*)(qb + 48 + hi * 8);
#pragma unroll
    for (int i = 0; i < 16; ++i) {
      Q1[i] = (__bf16)((float)Q1[i] * scale);
      Q2[i] = (__bf16)((float)Q2[i] * scale);
    }
  }

  float m1[8], l1[8], m2[8], l2[8];
  v8f O1[4], O2[4];
#pragma unroll
  for (int r = 0; r < 8; ++r) { m1[r] = m2[r] = -1e30f; l1[r] = l2[r] = 0.0f; }
#pragma unroll
  for (int t = 0; t < 4; ++t) { O1[t] = zero8(); O2[t] = zero8(); }

  auto softmax_update = [&](v8f& Sa, v8f& Sb, float* m, float* l, v8f* O) {
#pragma unroll
    for (int r = 0; r < 8; ++r) {
      const float mn = fmaxf(m[r], row_max16(fmaxf(Sa[r], Sb[r])));
      const float alpha = __expf(m[r] - mn);
      m[r] = mn;
      const float pa = __expf(Sa[r] - mn);
      const float pb = __expf(Sb[r] - mn);
      Sa[r] = pa; Sb[r] = pb;
      l[r] = l[r] * alpha + row_sum16(pa + pb);
#pragma unroll
      for (int t = 0; t < 4; ++t) O[t][r] *= alpha;
    }
  };

#if HAVE_TDM
  const unsigned klds_off = (unsigned)(unsigned long long)(const void*)Klds;
#endif

  for (int kb = 0; kb < SEQ / 32; ++kb) {
    __syncthreads();
#if HAVE_TDM
    if (wave == 0) {
      // TDM: 2D tile [32 rows x 64 bf16], row stride 3072 bf16, global->LDS.
      const unsigned long long ga = (unsigned long long)(const void*)(
          qkv + (long)(b * SEQ + kb * 32) * 3072 + 1024 + h * 64);
      v4u g0;
      g0[0] = 1u;                                   // count=1, user descriptor
      g0[1] = klds_off;                             // lds_addr (bytes)
      g0[2] = (unsigned)(ga & 0xFFFFFFFFu);         // global_addr[31:0]
      g0[3] = (unsigned)((ga >> 32) & 0x1FFFFFFu)   // global_addr[56:32]
              | (2u << 30);                         // type = 2 (image)
      v8i g1;
      g1[0] = 0x00010000;        // workgroup_mask=0, data_size=1 (2 bytes)
      g1[1] = (int)(64u << 16);  // tensor_dim0 = 64 (low half in bits 63:48)
      g1[2] = (int)(32u << 16);  // tensor_dim1 = 32 (bits 95:80)
      g1[3] = (int)(64u << 16);  // tile_dim0 = 64 (bits 127:112)
      g1[4] = 32;                // tile_dim1 = 32 (bits 143:128)
      g1[5] = 3072;              // tensor_dim0_stride (bits 191:160)
      g1[6] = 0;
      g1[7] = 0;
      v4i zg4 = {0, 0, 0, 0};
      v8i zg8 = {0, 0, 0, 0, 0, 0, 0, 0};
      // 6-arg toolchain variant: (g0, g1, g2, g3, g4, cpol)
      __builtin_amdgcn_tensor_load_to_lds(g0, g1, zg4, zg4, zg8, 0);
      __builtin_amdgcn_s_wait_tensorcnt(0);
    }
#endif
    {  // cooperative stage: V transposed (+ K when no TDM)
      const int key = tid >> 3;  // 0..31
      const int seg = tid & 7;   // 8 bf16 each
      const __bf16* base = qkv + (long)(b * SEQ + kb * 32 + key) * 3072 + h * 64;
#if !HAVE_TDM
      *(int4*)(Klds + key * 64 + seg * 8) = *(const int4*)(base + 1024 + seg * 8);
#endif
      int4 vv = *(const int4*)(base + 2048 + seg * 8);
      const __bf16* vh = (const __bf16*)&vv;
#pragma unroll
      for (int j = 0; j < 8; ++j) Vlds[(seg * 8 + j) * 32 + key] = vh[j];
      if (kb + 1 < SEQ / 32)  // L2 prefetch next key block
        __builtin_prefetch(base + 32 * 3072 + 1024, 0, 1);
    }
    __syncthreads();

    // S = Q * K^T for both streams (keys split in two 16-wide tiles)
    v8f S1a = wmma_bf16(Q1, frag_b(Klds + 0 * 64, 64), zero8());
    v8f S1b = wmma_bf16(Q1, frag_b(Klds + 16 * 64, 64), zero8());
    v8f S2a = wmma_bf16(Q2, frag_b(Klds + 0 * 64 + 32, 64), zero8());
    v8f S2b = wmma_bf16(Q2, frag_b(Klds + 16 * 64 + 32, 64), zero8());

    softmax_update(S1a, S1b, m1, l1, O1);
    softmax_update(S2a, S2b, m2, l2, O2);

    // Re-layout P (C layout) -> A layout via per-wave LDS staging
    __bf16* p1 = Pst[wave][0];
    __bf16* p2 = Pst[wave][1];
#pragma unroll
    for (int r = 0; r < 8; ++r) {
      const int rw = r + hi * 8;
      p1[rw * 32 + lo]      = (__bf16)S1a[r];
      p1[rw * 32 + 16 + lo] = (__bf16)S1b[r];
      p2[rw * 32 + lo]      = (__bf16)S2a[r];
      p2[rw * 32 + 16 + lo] = (__bf16)S2b[r];
    }
    const v16bf P1f = frag_a(p1, 32);
    const v16bf P2f = frag_a(p2, 32);
#pragma unroll
    for (int t = 0; t < 4; ++t) {
      const v16bf Vf = frag_b(Vlds + t * 16 * 32, 32);
      O1[t] = wmma_bf16(P1f, Vf, O1[t]);
      O2[t] = wmma_bf16(P2f, Vf, O2[t]);
    }
  }

  // Finalize: 1/l, diff, RMS-norm over 64, *(1 - lambda_init), store bf16
  float wreg[4];
#pragma unroll
  for (int t = 0; t < 4; ++t) wreg[t] = sub_w[t * 16 + lo];
#pragma unroll
  for (int r = 0; r < 8; ++r) {
    const float i1 = 1.0f / l1[r];
    const float i2 = lam / l2[r];
    float o[4];
    float ss = 0.0f;
#pragma unroll
    for (int t = 0; t < 4; ++t) {
      o[t] = O1[t][r] * i1 - O2[t][r] * i2;
      ss += o[t] * o[t];
    }
    ss = row_sum16(ss);
    const float rms = rsqrtf(ss * (1.0f / 64.0f) + RMS_EPS);
    const long n = q0 + r + hi * 8;
    __bf16* op = obuf + ((long)b * SEQ + n) * 1024 + h * 64;
#pragma unroll
    for (int t = 0; t < 4; ++t)
      op[t * 16 + lo] = (__bf16)(o[t] * rms * wreg[t] * 0.8f);
  }
}

// ---------------------------------------------------------------------------
extern "C" void kernel_launch(void* const* d_in, const int* in_sizes, int n_in,
                              void* d_out, int out_size, void* d_ws, size_t ws_size,
                              hipStream_t stream) {
  const float* x      = (const float*)d_in[0];
  const float* w_qkv  = (const float*)d_in[1];
  const float* w_proj = (const float*)d_in[2];
  const float* b_proj = (const float*)d_in[3];
  const float* lq1    = (const float*)d_in[4];
  const float* lk1    = (const float*)d_in[5];
  const float* lq2    = (const float*)d_in[6];
  const float* lk2    = (const float*)d_in[7];
  const float* snw    = (const float*)d_in[8];
  float* out = (float*)d_out;

  const long M = (long)BATCH * SEQ;  // 16384
  char* ws = (char*)d_ws;
  float*  lam  = (float*)ws;
  __bf16* qkv  = (__bf16*)(ws + 16);
  __bf16* obuf = (__bf16*)(ws + 16 + (size_t)M * 3072 * sizeof(__bf16));

  lambda_kernel<<<1, 32, 0, stream>>>(lq1, lk1, lq2, lk2, lam);

  dim3 g1(M / 128, 3072 / 128);
  gemm_wmma_kernel<false, false><<<g1, 256, 0, stream>>>(
      x, w_qkv, nullptr, qkv, (int)M, 3072, DIM);

  dim3 g2(SEQ / 128, BATCH * NH);
  diffattn_kernel<<<g2, 256, 0, stream>>>(qkv, lam, snw, obuf);

  dim3 g3(M / 128, DIM / 128);
  gemm_wmma_kernel<true, true><<<g3, 256, 0, stream>>>(
      obuf, w_proj, b_proj, out, (int)M, DIM, DIM);
}
